// SelfAttentionLayer_89867895701901
// MI455X (gfx1250) — compile-verified, matrix operand order checked by reading, not run on hardware
//
#include <hip/hip_runtime.h>

// GAT-style dense attention for MI455X (gfx1250, wave32, WMMA).
//   Wh = h@W; e = leaky(e_src[i]+e_dst[j]); P = softmax_rows(e); out = elu(P@Wh)
// K1: Wh via v_wmma_f32_16x16x32_f16; writes Wh(f16) pre-swizzled into the WMMA
//     B-operand lane layout (C/D lane layout == B lane layout -> free swizzle).
// K2: flash-style stream over j. P tile built directly in A-operand lane layout
//     in exp2 domain (2 add + 1 max + 1 v_exp per weight); row sums computed on
//     the matrix pipe via a 5th WMMA against an all-ones B operand.

typedef __attribute__((ext_vector_type(16))) _Float16 v16h;
typedef __attribute__((ext_vector_type(8)))  _Float16 v8h;
typedef __attribute__((ext_vector_type(8)))  float    v8f;

#define NROWS 8192
#define FIN   128
#define FOUT  64
#define LRELU 0.2f
#define L2E   1.4426950408889634f

// ---------------- Kernel 1: Wh = h@W (WMMA f16), e_src, e_dst, swizzled Wh ----
__global__ __launch_bounds__(256) void gat_k1(
    const float* __restrict__ h, const float* __restrict__ W,
    const float* __restrict__ a, _Float16* __restrict__ wsB,
    float* __restrict__ e_src, float* __restrict__ e_dst) {
  __shared__ _Float16 Wt[FOUT * FIN];  // W transposed, f16: Wt[col][k], 16 KB
  const int tid = threadIdx.x;
  for (int idx = tid; idx < FOUT * FIN; idx += 256) {
    const int col = idx >> 7, k = idx & (FIN - 1);
    Wt[idx] = (_Float16)W[k * FOUT + col];
  }
  __syncthreads();

  const int wave = tid >> 5;
  const int lane = tid & 31;
  const int r    = lane & 15;   // row within 16-row tile (A) / column (B,C)
  const int hf   = lane >> 4;   // K-half selector per ISA A/B 16-bit layout
  const int i0   = (blockIdx.x * 8 + wave) * 16;

  // A operands: 16x32 f16 tiles of h. Lane layout (ISA 7.12.2):
  //  lanes 0-15:  e[0..7]=K0..7,  e[8..15]=K16..23
  //  lanes 16-31: e[0..7]=K8..15, e[8..15]=K24..31
  v16h A[4];
  const float* hrow = h + (size_t)(i0 + r) * FIN;
#pragma unroll
  for (int kc = 0; kc < 4; ++kc) {
    const int b1 = kc * 32 + 8 * hf;
    float4 f0 = *(const float4*)(hrow + b1);
    float4 f1 = *(const float4*)(hrow + b1 + 4);
    float4 f2 = *(const float4*)(hrow + b1 + 16);
    float4 f3 = *(const float4*)(hrow + b1 + 20);
    v16h aa;
    aa[0]=(_Float16)f0.x;  aa[1]=(_Float16)f0.y;  aa[2]=(_Float16)f0.z;  aa[3]=(_Float16)f0.w;
    aa[4]=(_Float16)f1.x;  aa[5]=(_Float16)f1.y;  aa[6]=(_Float16)f1.z;  aa[7]=(_Float16)f1.w;
    aa[8]=(_Float16)f2.x;  aa[9]=(_Float16)f2.y;  aa[10]=(_Float16)f2.z; aa[11]=(_Float16)f2.w;
    aa[12]=(_Float16)f3.x; aa[13]=(_Float16)f3.y; aa[14]=(_Float16)f3.z; aa[15]=(_Float16)f3.w;
    A[kc] = aa;
  }

  const int jt = i0 >> 5;        // 32-row j-tile this 16-row tile belongs to
  const int ss = (i0 >> 4) & 1;  // which K-sub-half of that j-tile
  float ps[8], pd[8];
#pragma unroll
  for (int v = 0; v < 8; ++v) { ps[v] = 0.f; pd[v] = 0.f; }

#pragma unroll
  for (int ft = 0; ft < 4; ++ft) {
    v8f c = {};
#pragma unroll
    for (int kc = 0; kc < 4; ++kc) {
      // B operand: column N = ft*16+r of W; same K-run layout as A (symmetric).
      const _Float16* wb = &Wt[(ft * 16 + r) * FIN + kc * 32 + 8 * hf];
      v16h bb;
#pragma unroll
      for (int q = 0; q < 8; ++q) { bb[q] = wb[q]; bb[8 + q] = wb[16 + q]; }
      c = __builtin_amdgcn_wmma_f32_16x16x32_f16(false, A[kc], false, bb,
                                                 (short)0, c, false, false);
    }
    // C layout: lane holds Wh[i0 + 8*hf + v][ft*16 + r], v=0..7.
    const float a1v = a[ft * 16 + r];
    const float a2v = a[FOUT + ft * 16 + r];
    v8h hh;
#pragma unroll
    for (int v = 0; v < 8; ++v) {
      ps[v] = fmaf(c[v], a1v, ps[v]);
      pd[v] = fmaf(c[v], a2v, pd[v]);
      hh[v] = (_Float16)c[v];
    }
    // C/D lane layout == B-operand lane layout for the next GEMM: one 16B store.
    *(v8h*)(wsB + (((size_t)(jt * 4 + ft) * 32 + lane) * 16 + 8 * ss)) = hh;
  }

  // Reduce column-partials across the 16 lanes of each half-group.
#pragma unroll
  for (int v = 0; v < 8; ++v) {
    float x = ps[v];
    x += __shfl_xor(x, 1); x += __shfl_xor(x, 2);
    x += __shfl_xor(x, 4); x += __shfl_xor(x, 8);
    ps[v] = x;
    float y = pd[v];
    y += __shfl_xor(y, 1); y += __shfl_xor(y, 2);
    y += __shfl_xor(y, 4); y += __shfl_xor(y, 8);
    pd[v] = y;
  }
  if (r == 0) {
#pragma unroll
    for (int v = 0; v < 8; ++v) {
      e_src[i0 + 8 * hf + v] = ps[v];
      e_dst[i0 + 8 * hf + v] = pd[v];
    }
  }
}

// ---------------- Kernel 1.5: Dmax = max_j e_dst[j] -------------------------
__global__ __launch_bounds__(256) void gat_kmax(const float* __restrict__ e_dst,
                                                float* __restrict__ dmax) {
  __shared__ float red[256];
  float m = -3.4e38f;
  for (int j = threadIdx.x; j < NROWS; j += 256) m = fmaxf(m, e_dst[j]);
  red[threadIdx.x] = m;
  __syncthreads();
  for (int s = 128; s > 0; s >>= 1) {
    if (threadIdx.x < s) red[threadIdx.x] = fmaxf(red[threadIdx.x], red[threadIdx.x + s]);
    __syncthreads();
  }
  if (threadIdx.x == 0) *dmax = red[0];
}

// ---------------- Kernel 2: streaming softmax(P) @ Wh via WMMA, + ELU -------
__global__ __launch_bounds__(256) void gat_k2(
    const _Float16* __restrict__ wsB, const float* __restrict__ e_src,
    const float* __restrict__ e_dst, const float* __restrict__ dmax,
    float* __restrict__ out) {
  __shared__ float dA[NROWS];   // log2e * e_dst          (32 KB)
  __shared__ float dB[NROWS];   // log2e * 0.2 * e_dst    (32 KB)
  for (int j = threadIdx.x * 4; j < NROWS; j += 256 * 4) {
    float4 d = *(const float4*)&e_dst[j];
    float4 va = make_float4(d.x * L2E, d.y * L2E, d.z * L2E, d.w * L2E);
    float4 vb = make_float4(va.x * LRELU, va.y * LRELU, va.z * LRELU, va.w * LRELU);
    *(float4*)&dA[j] = va;
    *(float4*)&dB[j] = vb;
  }
  __syncthreads();

  const int wave = threadIdx.x >> 5;
  const int lane = threadIdx.x & 31;
  const int r    = lane & 15;
  const int hf   = lane >> 4;
  const int i0   = (blockIdx.x * 8 + wave) * 16;

  // leaky(s+d) = max(s+d, 0.2s+0.2d); row max m = leaky(s + max_j d) (monotone).
  const float s  = e_src[i0 + r];
  const float t0 = s + *dmax;
  const float m  = fmaxf(t0, LRELU * t0);
  const float sA = L2E * (s - m);          // exp(l-m) = exp2(max(sA+dA, sB+dB))
  const float sB = L2E * (LRELU * s - m);

  v16h ones;
#pragma unroll
  for (int q = 0; q < 16; ++q) ones[q] = (_Float16)1.0f;

  v8f c0 = {}, c1 = {}, c2 = {}, c3 = {}, c4 = {};

  const v16h* bbase = (const v16h*)wsB;
  for (int jt = 0; jt < NROWS / 32; ++jt) {
    // A-operand lane layout: lanes 0-15 K in {0-7,16-23}, lanes 16-31 K in {8-15,24-31}
    const float* pA = &dA[jt * 32 + 8 * hf];
    const float* pB = &dB[jt * 32 + 8 * hf];
    v16h P;
#pragma unroll
    for (int q = 0; q < 8; ++q) {
      float p1 = __builtin_amdgcn_exp2f(fmaxf(sA + pA[q],      sB + pB[q]));
      float p2 = __builtin_amdgcn_exp2f(fmaxf(sA + pA[16 + q], sB + pB[16 + q]));
      P[q]     = (_Float16)p1;
      P[8 + q] = (_Float16)p2;
    }
    const v16h* bt = bbase + (size_t)jt * 128 + lane;  // coalesced 32B/lane
    __builtin_prefetch(bt + 128, 0, 0);                // next j-tile -> global_prefetch
    v16h b0 = bt[0];
    v16h b1 = bt[32];
    v16h b2 = bt[64];
    v16h b3 = bt[96];
    c0 = __builtin_amdgcn_wmma_f32_16x16x32_f16(false, P, false, b0,   (short)0, c0, false, false);
    c1 = __builtin_amdgcn_wmma_f32_16x16x32_f16(false, P, false, b1,   (short)0, c1, false, false);
    c2 = __builtin_amdgcn_wmma_f32_16x16x32_f16(false, P, false, b2,   (short)0, c2, false, false);
    c3 = __builtin_amdgcn_wmma_f32_16x16x32_f16(false, P, false, b3,   (short)0, c3, false, false);
    // Row sums on the matrix pipe: every column of c4 = sum_j P[row][j],
    // and c4[v] lands in the same lane/register slot as cc[ft][v]'s row.
    c4 = __builtin_amdgcn_wmma_f32_16x16x32_f16(false, P, false, ones, (short)0, c4, false, false);
  }

  float inv[8];
#pragma unroll
  for (int v = 0; v < 8; ++v) inv[v] = 1.0f / c4[v];

  const v8f cc[4] = {c0, c1, c2, c3};
#pragma unroll
  for (int ft = 0; ft < 4; ++ft) {
#pragma unroll
    for (int v = 0; v < 8; ++v) {
      float x = cc[ft][v] * inv[v];
      float y = x > 0.f ? x : (__expf(x) - 1.f);  // ELU (alpha=1)
      out[(size_t)(i0 + 8 * hf + v) * FOUT + ft * 16 + r] = y;
    }
  }
}

// ---------------- Launch ----------------------------------------------------
extern "C" void kernel_launch(void* const* d_in, const int* in_sizes, int n_in,
                              void* d_out, int out_size, void* d_ws, size_t ws_size,
                              hipStream_t stream) {
  const float* h = (const float*)d_in[0];  // 8192 x 128
  const float* W = (const float*)d_in[1];  // 128 x 64
  const float* a = (const float*)d_in[2];  // 128 x 1
  float* out = (float*)d_out;              // 8192 x 64

  char* ws = (char*)d_ws;
  _Float16* wsB  = (_Float16*)ws;                                  // 1 MB swizzled Wh(f16)
  float*    e_s  = (float*)(ws + (size_t)NROWS * FOUT * 2);        // 32 KB
  float*    e_d  = e_s + NROWS;                                    // 32 KB
  float*    dmax = e_d + NROWS;                                    // 4 B

  gat_k1  <<<NROWS / 128, 256, 0, stream>>>(h, W, a, wsB, e_s, e_d);
  gat_kmax<<<1,           256, 0, stream>>>(e_d, dmax);
  gat_k2  <<<NROWS / 128, 256, 0, stream>>>(wsB, e_s, e_d, dmax, out);
}
